// ProjectionMatcher_40054865003039
// MI455X (gfx1250) — matrix operand order repaired and checked by reading, not compiled, not run on hardware
//
#include <hip/hip_runtime.h>
#include <hip/hip_bf16.h>
#include <math.h>

typedef __attribute__((ext_vector_type(16))) _Float16 v16h;
typedef __attribute__((ext_vector_type(8)))  float    v8f;
typedef __attribute__((ext_vector_type(4)))  unsigned v4u;
typedef __attribute__((ext_vector_type(8)))  int      v8i;
typedef __attribute__((ext_vector_type(4)))  int      v4i;

#define D_       128
#define W_       65
#define B_       16
#define P_       343
#define NPTS     (D_ * W_)              // 8320
#define PI64     0.04908738521234052f   // pi/64
#define INV128   0.0078125f
#define PRODSTR  130                    // 65 dwords/col -> conflict-free
#define PRODCOLS 66                     // 65 data cols + 1 zero dummy col
#define ZCOL     65                     // dummy zero column index
#define TMPSTR   98                     // 49 dwords/row -> conflict-free
#define NFRAG_E  16                     // (mt 2)*(kt 4)*(re/im)
#define NFRAG_C  12                     // (nt 2)*(kt 3)*(re/im)
#define EFRAG(kind, mt, kt) ((((mt)*4 + (kt))*2) + (kind))
#define CFRAG(kind, nt, kt) ((((nt)*3 + (kt))*2) + (kind))
#define NTHREADS 320                    // 10 wave32: one stage-B tile per wave

#if defined(__has_builtin)
# if __has_builtin(__builtin_amdgcn_tensor_load_to_lds) && __has_builtin(__builtin_amdgcn_s_wait_tensorcnt)
#  define USE_TDM 1
# endif
#endif
#ifndef USE_TDM
# define USE_TDM 0
#endif

// ---------------------------------------------------------------------------
// Kernel 0: precompute twiddle fragments as WMMA A/B operands, in this file's contiguous k order.
// Storage: tw[frag*256 + d*32 + lane], d = VGPR index (2 packed f16 per dword).
//   E frags (A, 16x32): elem(2d+e): kA = kt*32 + 16*half + 2*d + e
//       val = {cos,sin}(2pi*yraw*((kA+64)&127)/128)/128, yraw=(112+mt*16+lcol)&127
//   C frags (B, 32x16): elem(2d+e): k = kt*32 + 16*half + 2*d + e
//       val = {w*cos, -w*sin}(2pi*xraw*k/128)/128, w=0 (k>64) /1 (k==0,64) /2
// ---------------------------------------------------------------------------
__global__ __launch_bounds__(256) void k_twiddle(unsigned* __restrict__ tw) {
    int g = blockIdx.x * 256 + threadIdx.x;
    if (g >= (NFRAG_E + NFRAG_C) * 32) return;
    int frag = g >> 5;
    int lane = g & 31;
    int half = lane >> 4;
    int lcol = lane & 15;
    #pragma unroll
    for (int d = 0; d < 8; ++d) {
        union { _Float16 h[2]; unsigned u; } pk;
        #pragma unroll
        for (int e = 0; e < 2; ++e) {
            float val;
            if (frag < NFRAG_E) {
                int kind = frag & 1, q = frag >> 1;
                int mt = q >> 2, kt = q & 3;
                int yraw = (112 + mt * 16 + lcol) & 127;
                int kA = kt * 32 + (half << 4) + (d << 1) + e;
                int ph = (yraw * (kA + 64)) & 127;
                float s, c;
                __sincosf((float)ph * PI64, &s, &c);
                val = (kind ? s : c) * INV128;
            } else {
                int f = frag - NFRAG_E;
                int kind = f & 1, q = f >> 1;
                int nt = q / 3, kt = q - nt * 3;
                int xraw = (112 + nt * 16 + lcol) & 127;
                int k = kt * 32 + (half << 4) + (d << 1) + e;
                float w = (k > 64) ? 0.0f : ((k == 0 || k == 64) ? 1.0f : 2.0f);
                int ph = (xraw * k) & 127;
                float s, c;
                __sincosf((float)ph * PI64, &s, &c);
                val = (kind ? (-w * s) : (w * c)) * INV128;
            }
            pk.h[e] = (_Float16)val;
        }
        tw[frag * 256 + d * 32 + lane] = pk.u;
    }
}

// ---------------------------------------------------------------------------
// Kernel 1a: soft mask + row rDFT
// ---------------------------------------------------------------------------
__global__ __launch_bounds__(256) void k_rowfft(const float* __restrict__ parts,
                                                float2* __restrict__ X) {
    int g = blockIdx.x * 256 + threadIdx.x;
    if (g >= B_ * D_ * W_) return;
    int kx = g % W_;
    int t  = g / W_;
    int y  = t % D_;
    int b  = t / D_;
    float dy = (float)(y - 64);
    float accR = 0.f, accI = 0.f;
    const float* row = parts + (b * D_ + y) * D_;
    for (int x = 0; x < D_; ++x) {
        float dx = (float)(x - 64);
        float r  = sqrtf(dy * dy + dx * dx);
        float m  = fminf(fmaxf((64.0f - r) * (1.0f / 3.2f), 0.0f), 1.0f);
        float v  = row[x] * m;
        int   ph = (kx * x) & 127;
        float s, c;
        __sincosf((float)ph * PI64, &s, &c);
        accR += v * c;
        accI -= v * s;
    }
    X[(b * D_ + y) * W_ + kx] = make_float2(accR, accI);
}

// ---------------------------------------------------------------------------
// Kernel 1b: column DFT + fftshift(y) + CTF
// ---------------------------------------------------------------------------
__global__ __launch_bounds__(256) void k_colfft(const float2* __restrict__ X,
                                                const float* __restrict__ ctf,
                                                float2* __restrict__ pdft) {
    int g = blockIdx.x * 256 + threadIdx.x;
    if (g >= B_ * D_ * W_) return;
    int kx  = g % W_;
    int t   = g / W_;
    int row = t % D_;
    int b   = t / D_;
    int kk  = (row + 64) & 127;
    float accR = 0.f, accI = 0.f;
    for (int y = 0; y < D_; ++y) {
        float2 v = X[(b * D_ + y) * W_ + kx];
        int   ph = (kk * y) & 127;
        float s, c;
        __sincosf((float)ph * PI64, &s, &c);
        accR += v.x * c + v.y * s;
        accI += v.y * c - v.x * s;
    }
    float cf = ctf[(b * D_ + row) * W_ + kx];
    pdft[(b * D_ + row) * W_ + kx] = make_float2(accR * cf, accI * cf);
}

// ---------------------------------------------------------------------------
// Kernel 2: Fourier central-slice extraction (trilinear gather)
// ---------------------------------------------------------------------------
__global__ __launch_bounds__(256) void k_slices(const float* __restrict__ vol,
                                                const float* __restrict__ rot,
                                                float2* __restrict__ projs) {
    int idx = blockIdx.x * 256 + threadIdx.x;
    int p   = blockIdx.y;
    if (idx >= NPTS) return;
    int i = idx / W_;
    int j = idx - i * W_;
    float fy = (float)(i - 64) * (1.0f / 128.0f);
    float fx = (float)j * (1.0f / 128.0f);
    const float* R = rot + p * 9;
    float x = R[0] * fx + R[1] * fy;
    float y = R[3] * fx + R[4] * fy;
    float z = R[6] * fx + R[7] * fy;
    bool  conj = (x < 0.0f);
    float sg = conj ? -1.0f : 1.0f;
    x *= sg; y *= sg; z *= sg;
    float xi = x * 128.0f;
    float yi = y * 128.0f + 64.0f;
    float zi = z * 128.0f + 64.0f;
    float z0 = floorf(zi), y0 = floorf(yi), x0 = floorf(xi);
    float re = 0.f, im = 0.f;
    #pragma unroll
    for (int dz = 0; dz < 2; ++dz)
      #pragma unroll
      for (int dy = 0; dy < 2; ++dy)
        #pragma unroll
        for (int dx = 0; dx < 2; ++dx) {
            float zc = z0 + dz, yc = y0 + dy, xc = x0 + dx;
            float w = (1.0f - fabsf(zi - zc)) * (1.0f - fabsf(yi - yc)) * (1.0f - fabsf(xi - xc));
            bool inb = (zc >= 0.f) && (zc < 128.f) && (yc >= 0.f) && (yc < 128.f) &&
                       (xc >= 0.f) && (xc < 65.f);
            float wt = inb ? w : 0.0f;
            int zci = min(max((int)zc, 0), 127);
            int yci = min(max((int)yc, 0), 127);
            int xci = min(max((int)xc, 0), 64);
            int off = (zci * D_ + yci) * W_ + xci;
            re += vol[off] * wt;
            im += vol[D_ * D_ * W_ + off] * wt;
        }
    if (conj) im = -im;
    projs[p * NPTS + idx] = make_float2(re, im);
}

// ---------------------------------------------------------------------------
// Fragment loaders
// ---------------------------------------------------------------------------
__device__ inline v16h frag_ld(const unsigned* baseU, int frag, int lane) {
    union { v16h v; unsigned u[8]; } h;
    #pragma unroll
    for (int d = 0; d < 8; ++d)
        h.u[d] = baseU[frag * 256 + d * 32 + lane];
    return h.v;
}
// B (32x16 f16) from column-major LDS: lane col N, k = 16*half + 2*v.
// Unconditional: out-of-range columns are clamped to a zeroed dummy column.
__device__ inline v16h lds_b_frag(const _Float16* colbase, int kb, int half) {
    union { v16h v; unsigned u[8]; } h;
    const _Float16* p = colbase + kb + (half << 4);
    #pragma unroll
    for (int v2 = 0; v2 < 8; ++v2)
        h.u[v2] = *(const unsigned*)(p + (v2 << 1));
    return h.v;
}
// A (16x32 f16) from row-major LDS: lane row M, k = 16*half + 2*v
__device__ inline v16h lds_a_frag(const _Float16* base, int row, int stride, int kb, int half) {
    union { v16h v; unsigned u[8]; } h;
    const _Float16* p = base + row * stride + kb + (half << 4);
    #pragma unroll
    for (int v2 = 0; v2 < 8; ++v2) {
        int k = v2 << 1;
        h.u[v2] = *(const unsigned*)(p + k);
    }
    return h.v;
}

// ---------------------------------------------------------------------------
// Kernel 3: one workgroup (10 wave32) per (b,p) pair.
// ---------------------------------------------------------------------------
__global__ __launch_bounds__(NTHREADS) void k_corr(const float2* __restrict__ pdft,
                                                   const float2* __restrict__ projs,
                                                   const unsigned* __restrict__ tw,
                                                   float* __restrict__ out) {
    __shared__ _Float16 sProdR[PRODCOLS * PRODSTR];
    __shared__ _Float16 sProdI[PRODCOLS * PRODSTR];
    __shared__ _Float16 sTmpR[32 * TMPSTR];
    __shared__ _Float16 sTmpI[32 * TMPSTR];
    __shared__ unsigned sTwE[NFRAG_E * 256];       // 16 KB: E frags, staged via TDM
    __shared__ float    sRedV4[4];
    __shared__ int      sRedI4[4];

    const int tid  = threadIdx.x;
    const int wg   = blockIdx.x;
    const int b    = wg / P_;
    const int p    = wg - b * P_;
    const int lane = tid & 31;
    const int wave = tid >> 5;
    const int half = lane >> 4;
    const int lcol = lane & 15;

#if USE_TDM
    // ---- TDM: async 1-D copy of E-fragment table (16 KB) global -> LDS,
    // overlapped with stage A. data_size=8B, tensor/tile dim0 = 2048 units.
    if (wave == 0) {
        const unsigned NE = (NFRAG_E * 256u * 4u) / 8u;      // 2048 8-byte units
        unsigned lds_off = (unsigned)(size_t)(void*)&sTwE[0];
        unsigned long long ga = (unsigned long long)(size_t)tw;
        v4u g0;
        g0.x = 1u;                                            // count=1 (valid D#)
        g0.y = lds_off;                                       // lds_addr
        g0.z = (unsigned)(ga & 0xFFFFFFFFu);                  // global_addr[31:0]
        g0.w = (unsigned)((ga >> 32) & 0x1FFFFFFu) | (2u << 30); // addr[56:32], type=2
        v8i g1;
        g1[0] = (int)(3u << 16);                              // data_size = 8B
        g1[1] = (int)((NE & 0xFFFFu) << 16);                  // tensor_dim0[15:0]
        g1[2] = (int)((NE >> 16) | (1u << 16));               // tensor_dim0[31:16], tensor_dim1=1
        g1[3] = (int)(NE << 16);                              // tile_dim0 = NE
        g1[4] = 0;                                            // tile_dim1/2 unused
        g1[5] = (int)NE;                                      // tensor_dim0_stride[31:0]
        g1[6] = 0;
        g1[7] = 0;
        v4i g2 = {0, 0, 0, 0};
        v4i g3 = {0, 0, 0, 0};
#if defined(__clang_major__) && __clang_major__ >= 23
        v8i g4 = {0, 0, 0, 0, 0, 0, 0, 0};
        __builtin_amdgcn_tensor_load_to_lds(g0, g1, g2, g3, g4, 0);
#else
        __builtin_amdgcn_tensor_load_to_lds(g0, g1, g2, g3, 0);
#endif
    }
#else
    for (int i2 = tid; i2 < NFRAG_E * 256; i2 += NTHREADS) sTwE[i2] = tw[i2];
#endif

    // ---- Stage A: prod = pdft[b] * conj(projs[p]) -> f16 LDS (column-major)
    const float2* pa = pdft + b * NPTS;
    const float2* pc = projs + p * NPTS;
    for (int i2 = tid; i2 < NPTS; i2 += NTHREADS) {
        float2 a = pa[i2];
        float2 c = pc[i2];
        float pr = a.x * c.x + a.y * c.y;
        float pi = a.y * c.x - a.x * c.y;
        int ky = i2 / W_;
        int kx = i2 - ky * W_;
        int la = kx * PRODSTR + ky;
        sProdR[la] = (_Float16)pr;
        sProdI[la] = (_Float16)pi;
    }
    for (int i2 = tid; i2 < D_; i2 += NTHREADS) {          // zero dummy column
        sProdR[ZCOL * PRODSTR + i2] = (_Float16)0.0f;
        sProdI[ZCOL * PRODSTR + i2] = (_Float16)0.0f;
    }
    for (int i2 = tid; i2 < 32 * TMPSTR; i2 += NTHREADS) {
        sTmpR[i2] = (_Float16)0.0f;
        sTmpI[i2] = (_Float16)0.0f;
    }
#if USE_TDM
    if (wave == 0) __builtin_amdgcn_s_wait_tensorcnt(0);
#endif
    __syncthreads();

    // ---- Stage B: complex y-iDFT (K=128) onto 32 selected raw rows via WMMA.
    // 10 tiles, exactly one per wave32.
    if (wave < 10) {
        int t  = wave;
        int mt = t / 5;
        int nt = t - mt * 5;
        int col = nt * 16 + lcol;
        int colc = (col <= 64) ? col : ZCOL;               // clamp to zero column
        const _Float16* cbR = sProdR + colc * PRODSTR;
        const _Float16* cbI = sProdI + colc * PRODSTR;
        v8f accR = {};
        v8f accI = {};
        #pragma unroll
        for (int kt = 0; kt < 4; ++kt) {
            int kb = kt * 32;
            v16h prf = lds_b_frag(cbR, kb, half);
            v16h pif = lds_b_frag(cbI, kb, half);
            v16h er  = frag_ld(sTwE, EFRAG(0, mt, kt), lane);
            v16h ei  = frag_ld(sTwE, EFRAG(1, mt, kt), lane);
            v16h ein = -ei;
            accR = __builtin_amdgcn_wmma_f32_16x16x32_f16(false, er,  false, prf, (short)0, accR, false, false);
            accR = __builtin_amdgcn_wmma_f32_16x16x32_f16(false, ein, false, pif, (short)0, accR, false, false);
            accI = __builtin_amdgcn_wmma_f32_16x16x32_f16(false, er,  false, pif, (short)0, accI, false, false);
            accI = __builtin_amdgcn_wmma_f32_16x16x32_f16(false, ei,  false, prf, (short)0, accI, false, false);
        }
        #pragma unroll
        for (int r = 0; r < 8; ++r) {
            int row = mt * 16 + r + half * 8;
            int cc  = nt * 16 + lcol;
            sTmpR[row * TMPSTR + cc] = (_Float16)accR[r];
            sTmpI[row * TMPSTR + cc] = (_Float16)accI[r];
        }
    }
    __syncthreads();

    // ---- Stage C: x-inverse-rDFT onto 32 selected cols (Hermitian weights),
    // then max/argmax straight from accumulators (wave32 shuffle reduce).
    if (wave < 4) {
        int mt = wave >> 1;
        int nt = wave & 1;
        int arow = mt * 16 + lcol;
        const unsigned* twC = tw + NFRAG_E * 256;
        v8f acc = {};
        #pragma unroll
        for (int kt = 0; kt < 3; ++kt) {
            int kb = kt * 32;
            v16h ar = lds_a_frag(sTmpR, arow, TMPSTR, kb, half);
            v16h ai = lds_a_frag(sTmpI, arow, TMPSTR, kb, half);
            v16h cr = frag_ld(twC, CFRAG(0, nt, kt), lane);
            v16h ci = frag_ld(twC, CFRAG(1, nt, kt), lane);
            acc = __builtin_amdgcn_wmma_f32_16x16x32_f16(false, ar, false, cr, (short)0, acc, false, false);
            acc = __builtin_amdgcn_wmma_f32_16x16x32_f16(false, ai, false, ci, (short)0, acc, false, false);
        }
        float bv = -3.402823466e+38f;
        int   bi = 0;
        #pragma unroll
        for (int r = 0; r < 8; ++r) {               // r ascending -> idx ascending:
            int row = mt * 16 + r + half * 8;       // strict '>' keeps first max
            int col = nt * 16 + lcol;
            int idx = row * 32 + col;
            float v = acc[r];
            if (v > bv) { bv = v; bi = idx; }
        }
        #pragma unroll
        for (int o = 16; o > 0; o >>= 1) {
            float ov = __shfl_down(bv, o, 32);
            int   oi = __shfl_down(bi, o, 32);
            if (ov > bv || (ov == bv && oi < bi)) { bv = ov; bi = oi; }
        }
        if (lane == 0) { sRedV4[wave] = bv; sRedI4[wave] = bi; }
    }
    __syncthreads();

    if (tid == 0) {
        float bv = sRedV4[0];
        int   bi = sRedI4[0];
        #pragma unroll
        for (int w2 = 1; w2 < 4; ++w2) {
            float v = sRedV4[w2];
            int  i3 = sRedI4[w2];
            if (v > bv || (v == bv && i3 < bi)) { bv = v; bi = i3; }
        }
        out[wg] = bv;
        out[B_ * P_ + wg * 2 + 0] = (float)(48 + (bi & 31));   // sx
        out[B_ * P_ + wg * 2 + 1] = (float)(48 + (bi >> 5));   // sy
    }
}

// ---------------------------------------------------------------------------
extern "C" void kernel_launch(void* const* d_in, const int* in_sizes, int n_in,
                              void* d_out, int out_size, void* d_ws, size_t ws_size,
                              hipStream_t stream) {
    const float* vol   = (const float*)d_in[0];  // [2,128,128,65]
    const float* parts = (const float*)d_in[1];  // [16,128,128]
    const float* ctf   = (const float*)d_in[2];  // [16,128,65]
    const float* rot   = (const float*)d_in[3];  // [343,3,3]
    float* out = (float*)d_out;

    char* ws = (char*)d_ws;
    size_t off = 0;
    float2* projs = (float2*)(ws + off); off += (size_t)P_ * NPTS * sizeof(float2);
    float2* pdft  = (float2*)(ws + off); off += (size_t)B_ * NPTS * sizeof(float2);
    float2* X     = (float2*)(ws + off); off += (size_t)B_ * NPTS * sizeof(float2);
    unsigned* tw  = (unsigned*)(ws + off);                     // 28672 B

    k_twiddle<<<((NFRAG_E + NFRAG_C) * 32 + 255) / 256, 256, 0, stream>>>(tw);
    int n1 = B_ * D_ * W_;                                     // 133120
    k_rowfft<<<(n1 + 255) / 256, 256, 0, stream>>>(parts, X);
    k_colfft<<<(n1 + 255) / 256, 256, 0, stream>>>(X, ctf, pdft);
    k_slices<<<dim3((NPTS + 255) / 256, P_), 256, 0, stream>>>(vol, rot, projs);
    k_corr<<<B_ * P_, NTHREADS, 0, stream>>>(pdft, projs, tw, out);
}
